// CapsuleLayer_4612794876419
// MI455X (gfx1250) — matrix operand order repaired and checked by reading, not compiled, not run on hardware
//
#include <hip/hip_runtime.h>
#include <cstdint>

// ---------------------------------------------------------------------------
// CapsNet dynamic routing for MI455X (gfx1250, wave32, WMMA bf16 16x16x32).
//
// B=64, C=16, R=1152, IN=32, OUT=64.
// u[b,c,r,o] = sum_k x[b,r,k]*W[c,r,k,o]; 5 GEMM passes (~24 GFLOP bf16),
// u never materialized (302 MB) -- recomputed from operands that fit in L2.
//
//  * W and x converted fp32->bf16 ONCE, stored PRE-SWIZZLED in the exact
//    wave32 WMMA A/B operand layouts (ISA 7.12.2). Per-pass staging is a
//    contiguous 4 KB tile copy -> GLOBAL_LOAD_ASYNC_TO_LDS_B128 (ASYNCcnt)
//    with ping-pong LDS buffers, one barrier per r.
//  * W traffic per pass drops 151->75.5 MB (L2-resident for passes 2..5).
//
// ws layout: s[64K f32] | v[64K f32] | logits[1.15M f32] | lmax/lsum[1K f32]
//            | part[36*64K f32] | Wsw[37.7M bf16] | Xsw[2.36M bf16]  (~95 MB)
// ---------------------------------------------------------------------------

#define BB 64
#define CC 16
#define RR 1152
#define KK 32
#define OO 64
#define BCO (BB * CC * OO)          // 65536
#define BCR (BB * CC * RR)          // 1179648
#define RC_ACC 32
#define NCHUNK_ACC (RR / RC_ACC)    // 36
#define RC_LOG 16
#define NCHUNK_LOG (RR / RC_LOG)    // 72
#define TILE 2048                   // bf16 elems per (c,r) operand tile (4 KB)

typedef __attribute__((ext_vector_type(16))) __bf16 v16bf;
typedef __attribute__((ext_vector_type(8)))  float  v8f;
typedef __attribute__((ext_vector_type(4)))  int    v4i;

enum { MODE_UNIFORM = 0, MODE_LOGIT = 1, MODE_WSUM = 2 };

#if __has_builtin(__builtin_amdgcn_global_load_async_to_lds_b128)
#define HAS_ASYNC 1
#else
#define HAS_ASYNC 0
#endif

__device__ __forceinline__ unsigned short f2bf(float f) {
  // round-to-nearest-even fp32 -> bf16
  unsigned int u = __float_as_uint(f);
  u += 0x7FFFu + ((u >> 16) & 1u);
  return (unsigned short)(u >> 16);
}

// 16-byte global->LDS stage: async DMA on gfx1250, sync copy fallback.
// Builtin prototype (from hipcc diagnostic): arg0 = AS1 int4* (global src),
// arg1 = AS3 int4* (LDS dst), then imm offset, imm cpol.
__device__ __forceinline__ void stage16(const unsigned short* g,
                                        unsigned short* l) {
#if HAS_ASYNC
  __builtin_amdgcn_global_load_async_to_lds_b128(
      (__attribute__((address_space(1))) v4i*)g,
      (__attribute__((address_space(3))) v4i*)l, 0, 0);
#else
  *(uint4*)l = *(const uint4*)g;
#endif
}

__device__ __forceinline__ void wait_async0() {
#if HAS_ASYNC
#if __has_builtin(__builtin_amdgcn_s_wait_asynccnt)
  __builtin_amdgcn_s_wait_asynccnt(0);
#else
  asm volatile("s_wait_asynccnt 0x0" ::: "memory");
#endif
#endif
}

// ---------------------------------------------------------------------------
// One-time pre-swizzle: W fp32 [C,R,K,O] -> bf16 B-operand tiles
// Wsw[c][r][ntile][lane][elem].  Block = one (c,r) tile.
// ---------------------------------------------------------------------------
__global__ __launch_bounds__(256)
void swizzleW_kernel(const float* __restrict__ W, unsigned short* __restrict__ Wsw) {
  __shared__ __align__(16) unsigned short Bls[TILE];
  const int tid = threadIdx.x;
  const int r = blockIdx.x, c = blockIdx.y;
  const int we = tid * 8, wk = we >> 6, wn0 = we & 63;

  const float* wp = W + ((size_t)(c * RR + r) * KK + wk) * OO + wn0;
  float4 f0 = *(const float4*)(wp);
  float4 f1 = *(const float4*)(wp + 4);
  float f[8] = {f0.x, f0.y, f0.z, f0.w, f1.x, f1.y, f1.z, f1.w};
  const int hi = (wk >> 3) & 1, jj = (wk & 7) + 8 * (wk >> 4);   // B layout: lane half (k>>3)&1, elem (k&7)+8*(k>>4)
#pragma unroll
  for (int q = 0; q < 8; ++q) {
    int n = wn0 + q;
    Bls[(((n >> 4) * 32) + (n & 15) + 16 * hi) * 16 + jj] = f2bf(f[q]);
  }
  __syncthreads();
  *(uint4*)&Wsw[(size_t)(c * RR + r) * TILE + tid * 8] = *(uint4*)&Bls[tid * 8];
}

// One-time pre-swizzle: x fp32 [B,R,K] -> bf16 A-operand tiles Xsw[r][...]
__global__ __launch_bounds__(256)
void swizzleX_kernel(const float* __restrict__ x, unsigned short* __restrict__ Xsw) {
  const int tid = threadIdx.x;
  const int r = blockIdx.x;
  const int sb = tid >> 2, skq = (tid & 3) * 8;

  const float* xp = x + (size_t)(sb * RR + r) * KK + skq;
  float4 f0 = *(const float4*)(xp);
  float4 f1 = *(const float4*)(xp + 4);
  float f[8] = {f0.x, f0.y, f0.z, f0.w, f1.x, f1.y, f1.z, f1.w};
  unsigned short u[8];
#pragma unroll
  for (int q = 0; q < 8; ++q) u[q] = f2bf(f[q]);
  uint4 pk;
  pk.x = (unsigned)u[0] | ((unsigned)u[1] << 16);
  pk.y = (unsigned)u[2] | ((unsigned)u[3] << 16);
  pk.z = (unsigned)u[4] | ((unsigned)u[5] << 16);
  pk.w = (unsigned)u[6] | ((unsigned)u[7] << 16);
  // A layout: lane = (m&15)+16*((k>>3)&1); elem j = (k&7)+8*(k>>4)
  const int dst = ((sb >> 4) * 32 + (sb & 15) + 16 * ((skq >> 3) & 1)) * 16 +
                  (skq >> 4) * 8;
  *(uint4*)&Xsw[(size_t)r * TILE + dst] = pk;
}

// ---------------------------------------------------------------------------
// Fused GEMM + routing workhorse. Grid: (R/RC, C); 256 threads = 8 waves.
// Wave w: M-tile (w>>1)*16, N-half (w&1)*32 -> 2 WMMAs per r.
// Pipeline: [barrier] -> issue async stage for r+1 -> compute r.
// ---------------------------------------------------------------------------
template <int MODE, int RC>
__global__ __launch_bounds__(256)
void capsgemm_kernel(const unsigned short* __restrict__ Wsw,  // B tiles bf16
                     const unsigned short* __restrict__ Xsw,  // A tiles bf16
                     const float* __restrict__ x,             // fp32 (WSUM)
                     const float* __restrict__ v,             // [B,C,O] (LOGIT)
                     const float* __restrict__ lmax,          // [B*C] (WSUM)
                     const float* __restrict__ lsum,          // [B*C] (WSUM)
                     const float* __restrict__ logits_in,     // [B,C,R] (WSUM)
                     float* __restrict__ logits_out,          // [B,C,R] (LOGIT)
                     float* __restrict__ part)                // [NCHUNK][BCO]
{
  __shared__ __align__(16) unsigned short Bsm[2][TILE];  // ping-pong B
  __shared__ __align__(16) unsigned short Asm[2][TILE];  // ping-pong A
  __shared__ float lbuf[4][2][2][8];

  const int tid   = threadIdx.x;
  const int wave  = tid >> 5;
  const int lane  = tid & 31;
  const int chunk = blockIdx.x;
  const int c     = blockIdx.y;
  const int r0    = chunk * RC;

  const int mt    = wave >> 1;
  const int nt2   = wave & 1;
  const int nlo   = lane & 15;
  const int mhalf = lane >> 4;

  // per-thread 16-byte staging slices
  const unsigned short* bsrc = Wsw + ((size_t)(c * RR) + r0) * TILE + tid * 8;
  const unsigned short* asrc = Xsw + (size_t)r0 * TILE + tid * 8;

  // WSUM: A rebuilt from fp32 x with the softmax weight folded in
  const int sb = tid >> 2, skq = (tid & 3) * 8;
  float wmax = 0.f, wrz = 0.f;
  if (MODE == MODE_WSUM) {
    wmax = lmax[sb * CC + c];
    wrz  = 1.0f / lsum[sb * CC + c];
  }
  const int adst = ((sb >> 4) * 32 + (sb & 15) + 16 * ((skq >> 3) & 1)) * 16 +
                   (skq >> 4) * 8;

  float vv0[8], vv1[8];
  if (MODE == MODE_LOGIT) {
#pragma unroll
    for (int p = 0; p < 8; ++p) {
      int b = mt * 16 + p + 8 * mhalf;
      vv0[p] = v[(b * CC + c) * OO + nt2 * 32 + nlo];
      vv1[p] = v[(b * CC + c) * OO + nt2 * 32 + 16 + nlo];
    }
  }

  v8f acc0 = {};
  v8f acc1 = {};

  // prologue: stage r0
  stage16(bsrc, &Bsm[0][tid * 8]);
  if (MODE != MODE_WSUM) stage16(asrc, &Asm[0][tid * 8]);

  for (int rr = 0; rr < RC; ++rr) {
    const int cur = rr & 1, nxt = cur ^ 1;
    const int r = r0 + rr;

    if (MODE == MODE_WSUM) {  // sync A staging (weighted), before barrier
      const float* xp = x + (size_t)(sb * RR + r) * KK + skq;
      float4 f0 = *(const float4*)(xp);
      float4 f1 = *(const float4*)(xp + 4);
      float l = logits_in[(sb * CC + c) * RR + r];
      float w = __expf(l - wmax) * wrz;
      float f[8] = {f0.x * w, f0.y * w, f0.z * w, f0.w * w,
                    f1.x * w, f1.y * w, f1.z * w, f1.w * w};
      unsigned short u[8];
#pragma unroll
      for (int q = 0; q < 8; ++q) u[q] = f2bf(f[q]);
      uint4 pk;
      pk.x = (unsigned)u[0] | ((unsigned)u[1] << 16);
      pk.y = (unsigned)u[2] | ((unsigned)u[3] << 16);
      pk.z = (unsigned)u[4] | ((unsigned)u[5] << 16);
      pk.w = (unsigned)u[6] | ((unsigned)u[7] << 16);
      *(uint4*)&Asm[cur][adst] = pk;
    }

    wait_async0();       // stage for r complete (in-order ASYNCcnt)
    __syncthreads();     // staging visible; prior reads of [nxt] finished

    if (rr + 1 < RC) {   // issue next tile AFTER the barrier (no WAR race)
      stage16(bsrc + (size_t)(rr + 1) * TILE, &Bsm[nxt][tid * 8]);
      if (MODE != MODE_WSUM)
        stage16(asrc + (size_t)(rr + 1) * TILE, &Asm[nxt][tid * 8]);
    }

    // ---- WMMA: D = A(16x32 bf16) x B(32x16 bf16) + C(f32)
    v16bf a  = *reinterpret_cast<const v16bf*>(&Asm[cur][(mt * 32 + lane) * 16]);
    v16bf b0 = *reinterpret_cast<const v16bf*>(&Bsm[cur][((2 * nt2 + 0) * 32 + lane) * 16]);
    v16bf b1 = *reinterpret_cast<const v16bf*>(&Bsm[cur][((2 * nt2 + 1) * 32 + lane) * 16]);

    if (MODE == MODE_LOGIT) {
      v8f u0 = {}, u1 = {};
      u0 = __builtin_amdgcn_wmma_f32_16x16x32_bf16(false, a, false, b0,
                                                   (short)0, u0, false, false);
      u1 = __builtin_amdgcn_wmma_f32_16x16x32_bf16(false, a, false, b1,
                                                   (short)0, u1, false, false);
      float pl[8];
#pragma unroll
      for (int p = 0; p < 8; ++p) pl[p] = u0[p] * vv0[p] + u1[p] * vv1[p];
#pragma unroll
      for (int off = 8; off >= 1; off >>= 1) {
#pragma unroll
        for (int p = 0; p < 8; ++p) pl[p] += __shfl_xor(pl[p], off, 32);
      }
      if (lane == 0) {
#pragma unroll
        for (int p = 0; p < 8; ++p) lbuf[mt][nt2][0][p] = pl[p];
      }
      if (lane == 16) {
#pragma unroll
        for (int p = 0; p < 8; ++p) lbuf[mt][nt2][1][p] = pl[p];
      }
      __syncthreads();
      if (tid < 64) {  // one thread per b: combine N-halves, accumulate logit
        int b = tid;
        float l = lbuf[b >> 4][0][(b >> 3) & 1][b & 7] +
                  lbuf[b >> 4][1][(b >> 3) & 1][b & 7];
        logits_out[(size_t)(b * CC + c) * RR + r] += l;  // exclusive writer
      }
    } else {
      // K-reduction GEMM: accumulate across r in the C operand
      acc0 = __builtin_amdgcn_wmma_f32_16x16x32_bf16(false, a, false, b0,
                                                     (short)0, acc0, false, false);
      acc1 = __builtin_amdgcn_wmma_f32_16x16x32_bf16(false, a, false, b1,
                                                     (short)0, acc1, false, false);
    }
  }

  if (MODE != MODE_LOGIT) {
    // C/D layout: lane n = lane&15, M = p + 8*(lane>>4)
    float* pp = part + (size_t)chunk * BCO;
#pragma unroll
    for (int p = 0; p < 8; ++p) {
      int b = mt * 16 + p + 8 * mhalf;
      pp[(b * CC + c) * OO + nt2 * 32 + nlo]      = acc0[p];
      pp[(b * CC + c) * OO + nt2 * 32 + 16 + nlo] = acc1[p];
    }
  }
}

// ---------------------------------------------------------------------------
__global__ void zero_kernel(float* __restrict__ p, int n) {
  int i = blockIdx.x * blockDim.x + threadIdx.x;
  if (i < n) p[i] = 0.0f;
}

__global__ void reduce_part_kernel(const float* __restrict__ part,
                                   float* __restrict__ s, float scale) {
  int i = blockIdx.x * blockDim.x + threadIdx.x;
  if (i < BCO) {
    float acc = 0.0f;
    for (int ch = 0; ch < NCHUNK_ACC; ++ch) acc += part[(size_t)ch * BCO + i];
    s[i] = acc * scale;  // fixed-order sum: deterministic
  }
}

// per-(b,c) softmax statistics over r
__global__ void stats_kernel(const float* __restrict__ logits,
                             float* __restrict__ lmax,
                             float* __restrict__ lsum) {
  __shared__ float sm[128];
  const int bc  = blockIdx.x;
  const int tid = threadIdx.x;
  const float* l = logits + (size_t)bc * RR;

  float m = -3.4e38f;
  for (int i = tid; i < RR; i += 128) m = fmaxf(m, l[i]);
  sm[tid] = m;
  __syncthreads();
  for (int st = 64; st > 0; st >>= 1) {
    if (tid < st) sm[tid] = fmaxf(sm[tid], sm[tid + st]);
    __syncthreads();
  }
  const float mx = sm[0];
  __syncthreads();

  float z = 0.0f;
  for (int i = tid; i < RR; i += 128) z += __expf(l[i] - mx);
  sm[tid] = z;
  __syncthreads();
  for (int st = 64; st > 0; st >>= 1) {
    if (tid < st) sm[tid] += sm[tid + st];
    __syncthreads();
  }
  if (tid == 0) { lmax[bc] = mx; lsum[bc] = sm[0]; }
}

// squash: v = (|s|^2/(1+|s|^2)) * s/|s| per (b,c); 64 threads = 2 waves
__global__ void squash_kernel(const float* __restrict__ s,
                              float* __restrict__ dst) {
  __shared__ float red[2];
  const int bc = blockIdx.x;
  const int o  = threadIdx.x;
  float val = s[bc * OO + o];
  float e = val * val;
  for (int off = 16; off >= 1; off >>= 1) e += __shfl_xor(e, off, 32);
  if ((threadIdx.x & 31) == 0) red[threadIdx.x >> 5] = e;
  __syncthreads();
  float sq = red[0] + red[1];
  float sc = sq / ((1.0f + sq) * sqrtf(sq));
  dst[bc * OO + o] = val * sc;
}

// ---------------------------------------------------------------------------
extern "C" void kernel_launch(void* const* d_in, const int* in_sizes, int n_in,
                              void* d_out, int out_size, void* d_ws,
                              size_t ws_size, hipStream_t stream) {
  (void)in_sizes; (void)n_in; (void)out_size; (void)ws_size;

  const float* x = (const float*)d_in[0];   // [B,R,1,IN]
  const float* W = (const float*)d_in[1];   // [C,R,IN,OUT]
  float* out = (float*)d_out;               // [B,C,OUT]

  float* s      = (float*)d_ws;                       // BCO
  float* v      = s + BCO;                            // BCO
  float* logits = v + BCO;                            // BCR
  float* lmax   = logits + BCR;                       // B*C
  float* lsum   = lmax + BB * CC;                     // B*C
  float* part   = lsum + BB * CC;                     // NCHUNK_ACC*BCO
  unsigned short* Wsw = (unsigned short*)(part + (size_t)NCHUNK_ACC * BCO);
  unsigned short* Xsw = Wsw + (size_t)CC * RR * TILE; // [R][TILE]

  const dim3 gacc(NCHUNK_ACC, CC);
  const dim3 glog(NCHUNK_LOG, CC);

  // one-time bf16 pre-swizzle of both operands
  swizzleW_kernel<<<dim3(RR, CC), 256, 0, stream>>>(W, Wsw);
  swizzleX_kernel<<<RR, 256, 0, stream>>>(x, Xsw);
  zero_kernel<<<(BCR + 255) / 256, 256, 0, stream>>>(logits, BCR);

  // ---- iteration 0: uniform probs (1/R folded into the reduction)
  capsgemm_kernel<MODE_UNIFORM, RC_ACC><<<gacc, 256, 0, stream>>>(
      Wsw, Xsw, nullptr, nullptr, nullptr, nullptr, nullptr, nullptr, part);
  reduce_part_kernel<<<(BCO + 255) / 256, 256, 0, stream>>>(part, s,
                                                            1.0f / (float)RR);
  squash_kernel<<<BB * CC, OO, 0, stream>>>(s, v);

  // ---- iterations 1..2
  for (int it = 1; it <= 2; ++it) {
    capsgemm_kernel<MODE_LOGIT, RC_LOG><<<glog, 256, 0, stream>>>(
        Wsw, Xsw, nullptr, v, nullptr, nullptr, nullptr, logits, nullptr);
    stats_kernel<<<BB * CC, 128, 0, stream>>>(logits, lmax, lsum);
    capsgemm_kernel<MODE_WSUM, RC_ACC><<<gacc, 256, 0, stream>>>(
        Wsw, Xsw, x, nullptr, lmax, lsum, logits, nullptr, part);
    reduce_part_kernel<<<(BCO + 255) / 256, 256, 0, stream>>>(part, s, 1.0f);
    squash_kernel<<<BB * CC, OO, 0, stream>>>(s, (it == 2) ? out : v);
  }
}
